// NonLocalBlock_15066745275089
// MI455X (gfx1250) — matrix operand order repaired and checked by reading, not compiled, not run on hardware
//
#include <hip/hip_runtime.h>

// ---------------------------------------------------------------------------
// NonLocalBlock for MI455X (gfx1250), bf16 WMMA pipeline.
//   B=16, C=512, H=W=64 (N=4096), D=256
// Pipeline:
//   pack:   x, W_theta|W_phi|W_g (->Wcat[768,512]), W_final -> bf16
//   GEMM1:  tpg[b] = Wcat x xb[b]            [768,4096] bf16   (WMMA bf16)
//   GEMM2:  fraw[b] = theta[b] x phi_p[b]^T  [256,256]  fp32   (maxpool folded)
//   softmax over 255 cols -> fb bf16 (col 255 zeroed)
//   GEMM3:  yb[b] = fb[b] x g_p[b]           [256,4096] bf16
//   GEMM4:  out[b] = Wfin x yb[b] + x[b]     [512,4096] fp32
//
// A tile staged by the Tensor Data Mover (one tensor_load_to_lds per K-step,
// wave 0 only, s_wait_tensorcnt); B tile staged via packed-bf16 VALU
// (transpose/maxpool) or async global->LDS. 6-arg TDM builtin (clang-23).
// ---------------------------------------------------------------------------

#pragma clang diagnostic ignored "-Wdeprecated-builtins"

typedef __attribute__((ext_vector_type(16))) __bf16 v16bf;
typedef __attribute__((ext_vector_type(8)))  __bf16 v8bf;
typedef __attribute__((ext_vector_type(8)))  float  v8f;
typedef __attribute__((ext_vector_type(4)))  unsigned int u32x4;
typedef __attribute__((ext_vector_type(8)))  int i32x8;
typedef __attribute__((ext_vector_type(4)))  int i32x4;

#define BM 128
#define BN 128
#define BK 32

#if defined(__has_builtin)
#if __has_builtin(__builtin_amdgcn_tensor_load_to_lds)
#define HAVE_TDM 1
#endif
#endif

// Plain maxnum semantics (matches jnp.maximum on non-NaN data); the
// maximumnum variant expands with sNaN canonicalization and ~3x the VALU.
#define BF16MAX(a, b) __builtin_elementwise_max((a), (b))

struct Bf16x16 { uint4 lo, hi; };   // 32 bytes == 16 bf16

// Async copy of 32 contiguous bytes global -> LDS (two B128 transfers).
// INST_OFFSET applies to both the LDS and global addresses, so offset:16
// covers the second 16-byte chunk with the same address registers.
__device__ __forceinline__ void async_copy32B(unsigned lds_off, const void* src)
{
    asm volatile("global_load_async_to_lds_b128 %0, %1, off"
                 :: "v"(lds_off), "v"(src) : "memory");
    asm volatile("global_load_async_to_lds_b128 %0, %1, off offset:16"
                 :: "v"(lds_off), "v"(src) : "memory");
}

__device__ __forceinline__ void wait_async0()
{
    asm volatile("s_wait_asynccnt 0x0" ::: "memory");
}

__device__ __forceinline__ unsigned lds_offset(const void* p)
{
    return (unsigned)(unsigned long long)p;   // low 32 bits of flat LDS addr
}

// TDM 2-D tile load, bf16 elements (data_size=1 -> 2 bytes).
// Copies tile_d1 rows of tile_d0 elements from a row-major tensor with row
// stride `stride0` (elements) at `gaddr` into contiguous LDS at `lds_off`.
// Tensor dims are set to the tile dims (tile is always fully in-bounds here).
// D# layout per CDNA5 ISA ch.8 (groups 2/3 zero -> 2-D tensor).
__device__ __forceinline__ void tdm_load_tile_bf16(unsigned lds_off,
                                                   const void* gaddr,
                                                   unsigned tile_d0,
                                                   unsigned tile_d1,
                                                   unsigned long long stride0)
{
#ifdef HAVE_TDM
    const unsigned long long ga = (unsigned long long)gaddr;
    u32x4 g0;
    g0[0] = 1u;                                              // count=1, user D#
    g0[1] = lds_off;                                         // lds_addr
    g0[2] = (unsigned)(ga & 0xFFFFFFFFu);                    // global_addr lo
    g0[3] = (unsigned)((ga >> 32) & 0x1FFFFFFu) | (2u << 30);// addr hi | type=2

    i32x8 g1;
    g1[0] = (int)(1u << 16);                                 // data_size=1 (2B)
    g1[1] = (int)((tile_d0 & 0xFFFFu) << 16);                // tensor_dim0 lo16
    g1[2] = (int)(((tile_d0 >> 16) & 0xFFFFu) |              // tensor_dim0 hi
                  ((tile_d1 & 0xFFFFu) << 16));              // tensor_dim1 lo16
    g1[3] = (int)(((tile_d1 >> 16) & 0xFFFFu) |              // tensor_dim1 hi
                  ((tile_d0 & 0xFFFFu) << 16));              // tile_dim0
    g1[4] = (int)(tile_d1 & 0xFFFFu);                        // tile_dim1 (d2=0)
    g1[5] = (int)(stride0 & 0xFFFFFFFFull);                  // dim0_stride lo
    g1[6] = (int)((stride0 >> 32) & 0xFFFFull);              // dim0_stride hi
    g1[7] = 0;
    const i32x4 gz4 = {0, 0, 0, 0};
    const i32x8 gz8 = {0, 0, 0, 0, 0, 0, 0, 0};
    __builtin_amdgcn_tensor_load_to_lds(g0, g1, gz4, gz4, gz8, 0);
#else
    // Fallback: per-lane async copy (32B per thread of a 256-thread block).
    const int tid  = threadIdx.x;
    const int row  = tid >> 1;
    const int koff = (tid & 1) * 16;
    const __bf16* s = (const __bf16*)gaddr + (size_t)row * stride0 + koff;
    async_copy32B(lds_off + (unsigned)(row * tile_d0 + koff) * 2u, s);
#endif
}

// ---------------------------------------------------------------------------
// fp32 -> bf16 pack, 8 elems/thread
// ---------------------------------------------------------------------------
__global__ __launch_bounds__(256)
void pack_f32_bf16(const float* __restrict__ in, __bf16* __restrict__ out,
                   long long n)
{
    long long i = ((long long)blockIdx.x * 256 + threadIdx.x) * 8;
    if (i + 8 > n) return;
    float4 a = *(const float4*)(in + i);
    float4 b = *(const float4*)(in + i + 4);
    v8bf o;
    o[0] = (__bf16)a.x; o[1] = (__bf16)a.y; o[2] = (__bf16)a.z; o[3] = (__bf16)a.w;
    o[4] = (__bf16)b.x; o[5] = (__bf16)b.y; o[6] = (__bf16)b.z; o[7] = (__bf16)b.w;
    *(v8bf*)(out + i) = o;
}

// ---------------------------------------------------------------------------
// Tiled bf16 GEMM with fp32 accumulation on v_wmma_f32_16x16x32_bf16.
//
//  A  : [M][K] bf16 row-major (ld = K), staged by TDM.
//  B  : BMODE==0 : Bsrc is Bt [j][K] row-major (ld = K); rows j>=kn_valid -> 0
//       BMODE==1 : Bsrc is B  [K][N] row-major (ld = N); rows k>=kn_valid -> 0
//  MAXPOOL : B row is elementwise max with the next row (maxpool1d k=2,s=1)
//  EPI : 0 = bf16 store, 1 = fp32 store, 2 = fp32 + residual
//
//  Block = 256 threads = 8 waves; block tile 128x128; wave tile 32x64
//  (2x4 accumulators of 16x16). K stepped by 32.
// ---------------------------------------------------------------------------
template<int BMODE, bool MAXPOOL, int EPI>
__global__ __launch_bounds__(256)
void gemm_bf16_wmma(const __bf16* __restrict__ A,
                    const __bf16* __restrict__ Bsrc,
                    void* __restrict__ Cout,
                    const float* __restrict__ resid,
                    int M, int N, int K, int kn_valid,
                    long long sA, long long sB, long long sC, long long sR)
{
    __shared__ alignas(16) __bf16 As[BM * BK];   // [m][k]
    __shared__ alignas(16) __bf16 Bs[BN * BK];   // [n][k]  (K-contiguous)

    const int tid   = threadIdx.x;
    const int lane  = tid & 31;
    const int wid   = tid >> 5;
    const int waveM = wid >> 1;     // 0..3  -> 32 rows each
    const int waveN = wid & 1;      // 0..1  -> 64 cols each

    const int m0 = blockIdx.y * BM;
    const int n0 = blockIdx.x * BN;
    const int b  = blockIdx.z;

    A    += (size_t)b * (size_t)sA;
    Bsrc += (size_t)b * (size_t)sB;
    __bf16* Cb      = (__bf16*)Cout + (size_t)b * (size_t)sC;
    float*  Cf      = (float*)Cout + (size_t)b * (size_t)sC;
    const float* Rf = resid ? resid + (size_t)b * (size_t)sR : (const float*)0;

    const int ldB = (BMODE == 0) ? K : N;

    v8f acc[2][4];
#pragma unroll
    for (int mi = 0; mi < 2; ++mi)
#pragma unroll
        for (int ni = 0; ni < 4; ++ni)
            acc[mi][ni] = (v8f){0.f, 0.f, 0.f, 0.f, 0.f, 0.f, 0.f, 0.f};

    const uint4 zero4 = {0u, 0u, 0u, 0u};

    for (int k0 = 0; k0 < K; k0 += BK) {
        // ---------- stage A tile: one TDM transfer issued by wave 0 --------
#ifdef HAVE_TDM
        if (wid == 0)
            tdm_load_tile_bf16(lds_offset(As), A + (size_t)m0 * K + k0,
                               BK, BM, (unsigned long long)K);
#else
        tdm_load_tile_bf16(lds_offset(As), A + (size_t)m0 * K + k0,
                           BK, BM, (unsigned long long)K);
#endif
        // ---------- stage B tile into [n][k] -------------------------------
        if (BMODE == 0) {
            const int j    = tid >> 1;
            const int koff = (tid & 1) * 16;
            const int jg   = n0 + j;
            __bf16* d = &Bs[j * BK + koff];
            if (jg < kn_valid) {
                const __bf16* s = Bsrc + (size_t)jg * ldB + k0 + koff;
                if (MAXPOOL) {
                    const v8bf* sa = (const v8bf*)s;
                    const v8bf* sb = (const v8bf*)(s + ldB);
                    ((v8bf*)d)[0] = BF16MAX(sa[0], sb[0]);
                    ((v8bf*)d)[1] = BF16MAX(sa[1], sb[1]);
                } else {
                    async_copy32B(lds_offset(d), s);
                }
                if (k0 + BK < K) __builtin_prefetch((const void*)(s + BK), 0, 0);
            } else {
                *(uint4*)d       = zero4;
                *((uint4*)d + 1) = zero4;
            }
        } else {
            // transpose-stage: global [K][N] -> LDS [n][k]
            const int krow = tid >> 3;             // 0..31
            const int noff = (tid & 7) * 16;       // 0..112
            const int kg   = k0 + krow;
            alignas(16) __bf16 tmp[16];
            if (kg < kn_valid) {
                const __bf16* s = Bsrc + (size_t)kg * ldB + n0 + noff;
                if (MAXPOOL) {
                    const v8bf* sa = (const v8bf*)s;
                    const v8bf* sb = (const v8bf*)(s + ldB);
                    ((v8bf*)tmp)[0] = BF16MAX(sa[0], sb[0]);
                    ((v8bf*)tmp)[1] = BF16MAX(sa[1], sb[1]);
                } else {
                    *(uint4*)tmp       = *(const uint4*)s;
                    *((uint4*)tmp + 1) = *((const uint4*)s + 1);
                }
                if (k0 + BK < K)
                    __builtin_prefetch((const void*)(s + (size_t)BK * ldB), 0, 0);
            } else {
                *(uint4*)tmp       = zero4;
                *((uint4*)tmp + 1) = zero4;
            }
#pragma unroll
            for (int e = 0; e < 16; ++e)
                Bs[(noff + e) * BK + krow] = tmp[e];
        }
        // ---------- publish LDS tiles --------------------------------------
#ifdef HAVE_TDM
        if (wid == 0) __builtin_amdgcn_s_wait_tensorcnt(0);
        if (BMODE == 0 && !MAXPOOL) wait_async0();
#else
        wait_async0();
#endif
        __syncthreads();

        // ---------- fragments per documented 16-bit layouts ----------------
        // A 16x32: lane<16 holds row=lane, K in {kb..kb+7, kb+16..kb+23}, kb=(lane>>4)*8
        // B 32x16: lane holds col=lane&15, K in {kb..kb+7, kb+16..kb+23}, kb=(lane>>4)*8
        v16bf af[2], bfr[4];
#pragma unroll
        for (int mi = 0; mi < 2; ++mi) {
            const int row = waveM * 32 + mi * 16 + (lane & 15);
            const int kb  = (lane >> 4) * 8;
            const uint4* p = (const uint4*)&As[row * BK + kb];
            Bf16x16 t; t.lo = p[0]; t.hi = p[2];
            af[mi] = __builtin_bit_cast(v16bf, t);
        }
#pragma unroll
        for (int ni = 0; ni < 4; ++ni) {
            const int col = waveN * 64 + ni * 16 + (lane & 15);
            const int kb  = (lane >> 4) * 8;
            const uint4* p = (const uint4*)&Bs[col * BK + kb];
            Bf16x16 t; t.lo = p[0]; t.hi = p[2];
            bfr[ni] = __builtin_bit_cast(v16bf, t);
        }

#pragma unroll
        for (int mi = 0; mi < 2; ++mi)
#pragma unroll
            for (int ni = 0; ni < 4; ++ni)
                acc[mi][ni] = __builtin_amdgcn_wmma_f32_16x16x32_bf16(
                    false, af[mi], false, bfr[ni],
                    (short)0, acc[mi][ni], false, false);

        __syncthreads();
    }

    // ------------------ epilogue: C/D layout lanes0-15 M=r, lanes16-31 M=r+8
    const int hrow = (lane >> 4) * 8;
    const int ncol = lane & 15;
#pragma unroll
    for (int mi = 0; mi < 2; ++mi) {
#pragma unroll
        for (int ni = 0; ni < 4; ++ni) {
            const int gM = m0 + waveM * 32 + mi * 16 + hrow;
            const int gN = n0 + waveN * 64 + ni * 16 + ncol;
#pragma unroll
            for (int r = 0; r < 8; ++r) {
                const size_t off = (size_t)(gM + r) * N + gN;
                const float v = acc[mi][ni][r];
                if (EPI == 0)      Cb[off] = (__bf16)v;
                else if (EPI == 1) Cf[off] = v;
                else               Cf[off] = v + Rf[off];
            }
        }
    }
}

// ---------------------------------------------------------------------------
// Row softmax over the first 255 of 256 columns; one wave per row.
// Writes bf16, column 255 zeroed (it is padding for GEMM3's K).
// ---------------------------------------------------------------------------
__global__ __launch_bounds__(256)
void softmax_row255(const float* __restrict__ fraw, __bf16* __restrict__ fb)
{
    const int lane = threadIdx.x & 31;
    const int wid  = threadIdx.x >> 5;
    const long long row = (long long)blockIdx.x * 8 + wid;
    const float* rp = fraw + row * 256 + lane * 8;
    float4 v0 = *(const float4*)rp;
    float4 v1 = *((const float4*)rp + 1);
    float v[8] = {v0.x, v0.y, v0.z, v0.w, v1.x, v1.y, v1.z, v1.w};
    const int cbase = lane * 8;

    float m = -3.0e38f;
#pragma unroll
    for (int j = 0; j < 8; ++j)
        if (cbase + j < 255) m = fmaxf(m, v[j]);
#pragma unroll
    for (int off = 16; off > 0; off >>= 1)
        m = fmaxf(m, __shfl_xor(m, off, 32));

    float e[8];
    float s = 0.f;
#pragma unroll
    for (int j = 0; j < 8; ++j) {
        e[j] = (cbase + j < 255) ? __expf(v[j] - m) : 0.f;
        s += e[j];
    }
#pragma unroll
    for (int off = 16; off > 0; off >>= 1)
        s += __shfl_xor(s, off, 32);

    const float inv = 1.f / s;
    v8bf o;
#pragma unroll
    for (int j = 0; j < 8; ++j) o[j] = (__bf16)(e[j] * inv);
    *(v8bf*)(fb + row * 256 + lane * 8) = o;
}

// ---------------------------------------------------------------------------
extern "C" void kernel_launch(void* const* d_in, const int* in_sizes, int n_in,
                              void* d_out, int out_size, void* d_ws, size_t ws_size,
                              hipStream_t stream)
{
    (void)in_sizes; (void)n_in; (void)out_size; (void)ws_size;

    const float* x  = (const float*)d_in[0];   // [16,512,64,64]
    const float* Wt = (const float*)d_in[1];   // [256,512]
    const float* Wp = (const float*)d_in[2];   // [256,512]
    const float* Wg = (const float*)d_in[3];   // [256,512]
    const float* Wf = (const float*)d_in[4];   // [512,256]

    // Workspace layout (bytes). Region [0,67MB) holds xb during GEMM1, then is
    // reused for yb / fraw / fb (xb is dead after GEMM1). Total ~169 MB.
    char* ws = (char*)d_ws;
    __bf16* xb   = (__bf16*)(ws + 0LL);            // 16*512*4096 bf16 = 64 MiB
    __bf16* yb   = (__bf16*)(ws + 0LL);            // 16*256*4096 bf16 = 32 MiB
    float*  fraw = (float*) (ws + 33554432LL);     // 16*256*256 f32   =  4 MiB
    __bf16* fb   = (__bf16*)(ws + 50331648LL);     // 16*256*256 bf16  =  2 MiB
    __bf16* wcat = (__bf16*)(ws + 67108864LL);     // [768,512]
    __bf16* wfin = (__bf16*)(ws + 67895296LL);     // [512,256]
    __bf16* tpg  = (__bf16*)(ws + 68157440LL);     // 16*[768,4096] bf16 = 96 MiB

    // ---- pack to bf16 ----
    pack_f32_bf16<<<16384, 256, 0, stream>>>(x,  xb,             33554432LL);
    pack_f32_bf16<<<64,    256, 0, stream>>>(Wt, wcat + 0,       131072LL);
    pack_f32_bf16<<<64,    256, 0, stream>>>(Wp, wcat + 131072,  131072LL);
    pack_f32_bf16<<<64,    256, 0, stream>>>(Wg, wcat + 262144,  131072LL);
    pack_f32_bf16<<<64,    256, 0, stream>>>(Wf, wfin,           131072LL);

    // ---- GEMM1: tpg[b] = Wcat[768,512] x xb[b][512,4096]  (bf16 out) ----
    gemm_bf16_wmma<1, false, 0><<<dim3(32, 6, 16), 256, 0, stream>>>(
        wcat, xb, (void*)tpg, nullptr,
        768, 4096, 512, 512,
        0LL, 2097152LL, 3145728LL, 0LL);

    // ---- GEMM2: fraw[b] = theta[b][256,4096] x phi_p[b]^T (fp32 out) ----
    // Bt rows j = max(phi[j,:], phi[j+1,:]); rows >= 255 zeroed.
    gemm_bf16_wmma<0, true, 1><<<dim3(2, 2, 16), 256, 0, stream>>>(
        tpg, tpg + 1048576, (void*)fraw, nullptr,
        256, 256, 4096, 255,
        3145728LL, 3145728LL, 65536LL, 0LL);

    // ---- softmax over 255 cols -> fb (bf16, col 255 = 0) ----
    softmax_row255<<<512, 256, 0, stream>>>(fraw, fb);

    // ---- GEMM3: yb[b] = fb[b][256,256] x g_p[b][256,4096] (bf16 out) ----
    // B rows k = max(g[k,:], g[k+1,:]); row 255 zeroed (matches fb col 255 = 0).
    gemm_bf16_wmma<1, true, 0><<<dim3(32, 2, 16), 256, 0, stream>>>(
        fb, tpg + 2097152, (void*)yb, nullptr,
        256, 4096, 256, 255,
        65536LL, 3145728LL, 1048576LL, 0LL);

    // ---- GEMM4: out[b] = Wfin[512,256] x yb[b][256,4096] + x[b] (fp32) ----
    gemm_bf16_wmma<1, false, 2><<<dim3(32, 4, 16), 256, 0, stream>>>(
        wfin, yb, d_out, x,
        512, 4096, 256, 256,
        0LL, 1048576LL, 2097152LL, 2097152LL);
}
